// ReferenceModulatedCrossAttention_22840636080884
// MI455X (gfx1250) — matrix operand order repaired and checked by reading, not compiled, not run on hardware
//
#include <hip/hip_runtime.h>

#define HEADS 8
#define KDIM 1024
#define DMODEL 256
#define DHEAD 32
#define NB 8            // B*C
#define CATK 768
#define CATV 512
#define SIM_SCALE 0.17677669529663687f   // 32^-0.5

typedef __bf16 bf16_t;
typedef __attribute__((ext_vector_type(16))) __bf16 v16bf;
typedef __attribute__((ext_vector_type(8)))  __bf16 v8bf;
typedef __attribute__((ext_vector_type(8)))  float  v8f;

__device__ __forceinline__ v8f wmma_bf16(v16bf a, v16bf b, v8f c) {
  // 8 args: (neg_a, A, neg_b, B, c_mod, C, reuse_a, reuse_b)
  return __builtin_amdgcn_wmma_f32_16x16x32_bf16(false, a, false, b, (short)0, c, false, false);
}

// A-fragment (16x32 bf16, M x K). Per ISA 7.12.2: lane = hi*16 + m;
// element e<8  <-> K = e + 8*hi ; element e>=8 <-> K = 8 + e + 8*hi.
__device__ __forceinline__ v16bf load_afrag(const bf16_t* base, int stride,
                                            int row, int col, int lane) {
  const int hi = lane >> 4, rr = lane & 15;
  const bf16_t* p = base + (size_t)(row + rr) * (size_t)stride + col;
  v8bf lo = *(const v8bf*)(p + hi * 8);
  v8bf hh = *(const v8bf*)(p + 16 + hi * 8);
  return __builtin_shufflevector(lo, hh, 0,1,2,3,4,5,6,7,8,9,10,11,12,13,14,15);
}

// B-fragment (32x16 bf16, K x N), loaded from B^T stored row-major (N x K).
// lane = hi*16 + n ; element e<8 <-> K = e + 8*hi ; element e>=8 <-> K = 8 + e + 8*hi.
__device__ __forceinline__ v16bf load_bfrag(const bf16_t* baseT, int stride,
                                            int nrow, int kcol, int lane) {
  const int hi = lane >> 4, rr = lane & 15;
  const bf16_t* p = baseT + (size_t)(nrow + rr) * (size_t)stride + kcol;
  v8bf lo = *(const v8bf*)(p + hi * 8);
  v8bf hh = *(const v8bf*)(p + 16 + hi * 8);
  return __builtin_shufflevector(lo, hh, 0,1,2,3,4,5,6,7,8,9,10,11,12,13,14,15);
}

// ---------------- prep: concat inputs -> bf16, transpose weights -> bf16 ----
__global__ void prep_cat_kernel(const float* __restrict__ x,
                                const float* __restrict__ ci,
                                const float* __restrict__ rs,
                                bf16_t* __restrict__ catK) {
  size_t i = (size_t)blockIdx.x * blockDim.x + threadIdx.x;
  const size_t total = (size_t)NB * KDIM * CATK;
  const size_t step = (size_t)gridDim.x * blockDim.x;
  for (; i < total; i += step) {
    int c = (int)(i % CATK);
    size_t rk = i / CATK;              // bc*KDIM + k
    int k = (int)(rk % KDIM);
    int bc = (int)(rk / KDIM);
    float v;
    if (c < 256)        v = x [rk * 256 + c];
    else if (c < 512)   v = ci[rk * 256 + (c - 256)];
    else                v = rs[((size_t)(bc >> 2) * KDIM + k) * 256 + (c - 512)];
    catK[i] = (bf16_t)v;
  }
}

__global__ void prep_wt_kernel(const float* __restrict__ Wk,
                               const float* __restrict__ Wv,
                               const float* __restrict__ Wo,
                               bf16_t* __restrict__ WkT,
                               bf16_t* __restrict__ WvT,
                               bf16_t* __restrict__ WoT) {
  int i = blockIdx.x * blockDim.x + threadIdx.x;
  const int tot = 256 * 768 + 256 * 512 + 256 * 256;
  const int step = gridDim.x * blockDim.x;
  for (; i < tot; i += step) {
    if (i < 256 * 768) {
      int n = i / 768, k = i % 768;
      WkT[i] = (bf16_t)Wk[k * 256 + n];
    } else if (i < 256 * 768 + 256 * 512) {
      int j = i - 256 * 768;
      int n = j / 512, k = j % 512;
      WvT[j] = (bf16_t)Wv[k * 256 + n];
    } else {
      int j = i - 256 * 768 - 256 * 512;
      int n = j / 256, k = j % 256;
      WoT[j] = (bf16_t)Wo[k * 256 + n];
    }
  }
}

// ---------------- projections: cond = catK@Wk, refv = catV@Wv --------------
__global__ __launch_bounds__(256)
void proj_kernel(const bf16_t* __restrict__ catK,
                 const bf16_t* __restrict__ WkT, const bf16_t* __restrict__ WvT,
                 bf16_t* __restrict__ cond, bf16_t* __restrict__ condT,
                 bf16_t* __restrict__ refv, bf16_t* __restrict__ refvT) {
  const int bc = blockIdx.x >> 6;
  const int k0 = (blockIdx.x & 63) * 16;
  const int w = threadIdx.x >> 5;          // output column group (32 cols)
  const int lane = threadIdx.x & 31;
  const int hi = lane >> 4, n = lane & 15;
  const bf16_t* A = catK + (size_t)bc * KDIM * CATK;
  v8f zero = {};

  // cond = catK(16x768) @ Wk(768x256), this wave: cols [w*32, w*32+32)
  v8f c0 = zero, c1 = zero;
  for (int kc = 0; kc < 24; ++kc) {
    v16bf a  = load_afrag(A,   CATK, k0,        kc * 32, lane);
    v16bf b0 = load_bfrag(WkT, CATK, w * 32,      kc * 32, lane);
    v16bf b1 = load_bfrag(WkT, CATK, w * 32 + 16, kc * 32, lane);
    c0 = wmma_bf16(a, b0, c0);
    c1 = wmma_bf16(a, b1, c1);
  }
  {
    bf16_t* C  = cond  + (size_t)bc * KDIM * DMODEL;
    bf16_t* CT = condT + (size_t)bc * DMODEL * KDIM;
    for (int e = 0; e < 8; ++e) {
      int row = k0 + e + 8 * hi;
      int col = w * 32 + n;
      C [(size_t)row * DMODEL + col]        = (bf16_t)c0[e];
      C [(size_t)row * DMODEL + col + 16]   = (bf16_t)c1[e];
      CT[(size_t)col * KDIM + row]          = (bf16_t)c0[e];
      CT[(size_t)(col + 16) * KDIM + row]   = (bf16_t)c1[e];
    }
  }

  // refv = catV(16x512) @ Wv(512x256); catV = catK cols [0,256) U [512,768)
  v8f r0 = zero, r1 = zero;
  for (int kc = 0; kc < 16; ++kc) {
    int sc = (kc < 8) ? kc * 32 : 256 + kc * 32;
    v16bf a  = load_afrag(A,   CATK, k0,        sc,      lane);
    v16bf b0 = load_bfrag(WvT, CATV, w * 32,      kc * 32, lane);
    v16bf b1 = load_bfrag(WvT, CATV, w * 32 + 16, kc * 32, lane);
    r0 = wmma_bf16(a, b0, r0);
    r1 = wmma_bf16(a, b1, r1);
  }
  {
    bf16_t* R  = refv  + (size_t)bc * KDIM * DMODEL;
    bf16_t* RT = refvT + (size_t)bc * DMODEL * KDIM;
    for (int e = 0; e < 8; ++e) {
      int row = k0 + e + 8 * hi;
      int col = w * 32 + n;
      R [(size_t)row * DMODEL + col]        = (bf16_t)r0[e];
      R [(size_t)row * DMODEL + col + 16]   = (bf16_t)r1[e];
      RT[(size_t)col * KDIM + row]          = (bf16_t)r0[e];
      RT[(size_t)(col + 16) * KDIM + row]   = (bf16_t)r1[e];
    }
  }
}

// ------------- out = ((th_w mix . rowsoftmax(sim)) @ refv) @ Wo + bo -------
__global__ __launch_bounds__(256)
void attn_out_kernel(const bf16_t* __restrict__ cond,
                     const bf16_t* __restrict__ refv,
                     const bf16_t* __restrict__ refvT,
                     const bf16_t* __restrict__ WoT,
                     const float* __restrict__ bo,
                     const float* __restrict__ th_w,
                     float* __restrict__ out) {
  __shared__ alignas(16) float  ex[HEADS][16][32];
  __shared__ alignas(16) bf16_t mixA[HEADS][16][32];
  __shared__ alignas(16) bf16_t merged[16][DMODEL];

  const int bc = blockIdx.x >> 6;
  const int p0 = (blockIdx.x & 63) * 16;
  const int h = threadIdx.x >> 5;          // wave = head h (also out-group g)
  const int lane = threadIdx.x & 31;
  const int hi = lane >> 4, n = lane & 15;
  v8f zero = {};

  const bf16_t* C  = cond  + (size_t)bc * KDIM * DMODEL;
  const bf16_t* R  = refv  + (size_t)bc * KDIM * DMODEL;
  const bf16_t* RT = refvT + (size_t)bc * DMODEL * KDIM;

  float thr[HEADS];
  for (int j = 0; j < HEADS; ++j) thr[j] = th_w[h * HEADS + j];

  // A = cond rows [p0,p0+16), K = this head's 32 dims (one WMMA K-step)
  v16bf a = load_afrag(C, DMODEL, p0, h * DHEAD, lane);

  // pass 1: row max of sim over q
  float rm[8];
  for (int e = 0; e < 8; ++e) rm[e] = -3.0e38f;
  for (int qt = 0; qt < 64; ++qt) {
    v16bf b = load_bfrag(R, DMODEL, qt * 16, h * DHEAD, lane);
    v8f s = wmma_bf16(a, b, zero);
    for (int e = 0; e < 8; ++e) rm[e] = fmaxf(rm[e], s[e] * SIM_SCALE);
  }
  for (int e = 0; e < 8; ++e)
    for (int m = 1; m < 16; m <<= 1)
      rm[e] = fmaxf(rm[e], __shfl_xor(rm[e], m, 32));

  // pass 2: row sum of exp
  float rs[8];
  for (int e = 0; e < 8; ++e) rs[e] = 0.0f;
  for (int qt = 0; qt < 64; ++qt) {
    v16bf b = load_bfrag(R, DMODEL, qt * 16, h * DHEAD, lane);
    v8f s = wmma_bf16(a, b, zero);
    for (int e = 0; e < 8; ++e) rs[e] += __expf(s[e] * SIM_SCALE - rm[e]);
  }
  for (int e = 0; e < 8; ++e)
    for (int m = 1; m < 16; m <<= 1)
      rs[e] += __shfl_xor(rs[e], m, 32);
  float ri[8];
  for (int e = 0; e < 8; ++e) ri[e] = 1.0f / rs[e];

  // pass 3: attn -> talking-heads mix -> accumulate @ refv
  v8f acc0 = zero, acc1 = zero;
  for (int qc = 0; qc < 32; ++qc) {
    for (int t = 0; t < 2; ++t) {
      v16bf b = load_bfrag(R, DMODEL, qc * 32 + t * 16, h * DHEAD, lane);
      v8f s = wmma_bf16(a, b, zero);
      for (int e = 0; e < 8; ++e)
        ex[h][e + 8 * hi][t * 16 + n] = __expf(s[e] * SIM_SCALE - rm[e]) * ri[e];
    }
    __syncthreads();
    for (int t = 0; t < 2; ++t)
      for (int e = 0; e < 8; ++e) {
        float v = 0.0f;
        for (int j = 0; j < HEADS; ++j) v += thr[j] * ex[j][e + 8 * hi][t * 16 + n];
        mixA[h][e + 8 * hi][t * 16 + n] = (bf16_t)v;
      }
    __syncthreads();
    v16bf am = load_afrag(&mixA[h][0][0], 32, 0, 0, lane);
    v16bf b0 = load_bfrag(RT, KDIM, h * DHEAD,      qc * 32, lane);
    v16bf b1 = load_bfrag(RT, KDIM, h * DHEAD + 16, qc * 32, lane);
    acc0 = wmma_bf16(am, b0, acc0);
    acc1 = wmma_bf16(am, b1, acc1);
    __syncthreads();
  }

  // epilogue: merged(16x256) @ Wo + bo
  for (int e = 0; e < 8; ++e) {
    merged[e + 8 * hi][h * DHEAD + n]      = (bf16_t)acc0[e];
    merged[e + 8 * hi][h * DHEAD + 16 + n] = (bf16_t)acc1[e];
  }
  __syncthreads();
  v8f f0 = zero, f1 = zero;
  for (int kc = 0; kc < 8; ++kc) {
    v16bf am = load_afrag(&merged[0][0], DMODEL, 0, kc * 32, lane);
    v16bf b0 = load_bfrag(WoT, DMODEL, h * 32,      kc * 32, lane);
    v16bf b1 = load_bfrag(WoT, DMODEL, h * 32 + 16, kc * 32, lane);
    f0 = wmma_bf16(am, b0, f0);
    f1 = wmma_bf16(am, b1, f1);
  }
  float* O = out + (size_t)bc * KDIM * DMODEL;
  for (int e = 0; e < 8; ++e) {
    int row = p0 + e + 8 * hi;
    int col = h * 32 + n;
    O[(size_t)row * DMODEL + col]      = f0[e] + bo[col];
    O[(size_t)row * DMODEL + col + 16] = f1[e] + bo[col + 16];
  }
}

// --------- context_out = (cth_w mix . colsoftmax(sim))^T @ cond ------------
// Column softmax of sim == row softmax of simT; simT[q,p] = refv[q].cond[p].
__global__ __launch_bounds__(256)
void attn_ctx_kernel(const bf16_t* __restrict__ cond,
                     const bf16_t* __restrict__ condT,
                     const bf16_t* __restrict__ refv,
                     const float* __restrict__ cth_w,
                     float* __restrict__ ctx_out) {
  __shared__ alignas(16) float  ex[HEADS][16][32];
  __shared__ alignas(16) bf16_t mixA[HEADS][16][32];

  const int bc = blockIdx.x >> 6;
  const int q0 = (blockIdx.x & 63) * 16;
  const int h = threadIdx.x >> 5;
  const int lane = threadIdx.x & 31;
  const int hi = lane >> 4, n = lane & 15;
  v8f zero = {};

  const bf16_t* C  = cond  + (size_t)bc * KDIM * DMODEL;
  const bf16_t* CT = condT + (size_t)bc * DMODEL * KDIM;
  const bf16_t* R  = refv  + (size_t)bc * KDIM * DMODEL;

  float thr[HEADS];
  for (int j = 0; j < HEADS; ++j) thr[j] = cth_w[h * HEADS + j];

  // A = refv rows [q0,q0+16) -> simT tile rows are q, cols are p
  v16bf a = load_afrag(R, DMODEL, q0, h * DHEAD, lane);

  // pass 1: per-q max over p (== column max of sim)
  float rm[8];
  for (int e = 0; e < 8; ++e) rm[e] = -3.0e38f;
  for (int pt = 0; pt < 64; ++pt) {
    v16bf b = load_bfrag(C, DMODEL, pt * 16, h * DHEAD, lane);
    v8f s = wmma_bf16(a, b, zero);
    for (int e = 0; e < 8; ++e) rm[e] = fmaxf(rm[e], s[e] * SIM_SCALE);
  }
  for (int e = 0; e < 8; ++e)
    for (int m = 1; m < 16; m <<= 1)
      rm[e] = fmaxf(rm[e], __shfl_xor(rm[e], m, 32));

  // pass 2: per-q sum of exp over p
  float rs[8];
  for (int e = 0; e < 8; ++e) rs[e] = 0.0f;
  for (int pt = 0; pt < 64; ++pt) {
    v16bf b = load_bfrag(C, DMODEL, pt * 16, h * DHEAD, lane);
    v8f s = wmma_bf16(a, b, zero);
    for (int e = 0; e < 8; ++e) rs[e] += __expf(s[e] * SIM_SCALE - rm[e]);
  }
  for (int e = 0; e < 8; ++e)
    for (int m = 1; m < 16; m <<= 1)
      rs[e] += __shfl_xor(rs[e], m, 32);
  float ri[8];
  for (int e = 0; e < 8; ++e) ri[e] = 1.0f / rs[e];

  // pass 3: cattn^T -> mix -> accumulate @ cond
  v8f acc0 = zero, acc1 = zero;
  for (int pc = 0; pc < 32; ++pc) {
    for (int t = 0; t < 2; ++t) {
      v16bf b = load_bfrag(C, DMODEL, pc * 32 + t * 16, h * DHEAD, lane);
      v8f s = wmma_bf16(a, b, zero);
      for (int e = 0; e < 8; ++e)
        ex[h][e + 8 * hi][t * 16 + n] = __expf(s[e] * SIM_SCALE - rm[e]) * ri[e];
    }
    __syncthreads();
    for (int t = 0; t < 2; ++t)
      for (int e = 0; e < 8; ++e) {
        float v = 0.0f;
        for (int j = 0; j < HEADS; ++j) v += thr[j] * ex[j][e + 8 * hi][t * 16 + n];
        mixA[h][e + 8 * hi][t * 16 + n] = (bf16_t)v;
      }
    __syncthreads();
    v16bf am = load_afrag(&mixA[h][0][0], 32, 0, 0, lane);
    v16bf b0 = load_bfrag(CT, KDIM, h * DHEAD,      pc * 32, lane);
    v16bf b1 = load_bfrag(CT, KDIM, h * DHEAD + 16, pc * 32, lane);
    acc0 = wmma_bf16(am, b0, acc0);
    acc1 = wmma_bf16(am, b1, acc1);
    __syncthreads();
  }

  float* O = ctx_out + (size_t)bc * KDIM * DMODEL;
  for (int e = 0; e < 8; ++e) {
    int row = q0 + e + 8 * hi;
    int col = h * 32 + n;
    O[(size_t)row * DMODEL + col]      = acc0[e];
    O[(size_t)row * DMODEL + col + 16] = acc1[e];
  }
}

extern "C" void kernel_launch(void* const* d_in, const int* in_sizes, int n_in,
                              void* d_out, int out_size, void* d_ws, size_t ws_size,
                              hipStream_t stream) {
  const float* x    = (const float*)d_in[0];
  const float* ci   = (const float*)d_in[1];
  const float* rs   = (const float*)d_in[2];
  // d_in[3] = Wq — dead in reference, skipped
  const float* Wk   = (const float*)d_in[4];
  const float* Wv   = (const float*)d_in[5];
  const float* Wo   = (const float*)d_in[6];
  const float* bo   = (const float*)d_in[7];
  const float* thw  = (const float*)d_in[8];
  const float* cthw = (const float*)d_in[9];
  float* out = (float*)d_out;

  char* ws = (char*)d_ws;
  bf16_t* catK = (bf16_t*)ws;  ws += (size_t)NB * KDIM * CATK * sizeof(bf16_t);
  bf16_t* WkT  = (bf16_t*)ws;  ws += (size_t)256 * 768 * sizeof(bf16_t);
  bf16_t* WvT  = (bf16_t*)ws;  ws += (size_t)256 * 512 * sizeof(bf16_t);
  bf16_t* WoT  = (bf16_t*)ws;  ws += (size_t)256 * 256 * sizeof(bf16_t);
  bf16_t* cond = (bf16_t*)ws;  ws += (size_t)NB * KDIM * DMODEL * sizeof(bf16_t);
  bf16_t* condT= (bf16_t*)ws;  ws += (size_t)NB * DMODEL * KDIM * sizeof(bf16_t);
  bf16_t* refv = (bf16_t*)ws;  ws += (size_t)NB * KDIM * DMODEL * sizeof(bf16_t);
  bf16_t* refvT= (bf16_t*)ws;  ws += (size_t)NB * DMODEL * KDIM * sizeof(bf16_t);

  prep_cat_kernel<<<2048, 256, 0, stream>>>(x, ci, rs, catK);
  prep_wt_kernel<<<512, 256, 0, stream>>>(Wk, Wv, Wo, WkT, WvT, WoT);
  proj_kernel<<<NB * 64, 256, 0, stream>>>(catK, WkT, WvT, cond, condT, refv, refvT);
  attn_out_kernel<<<NB * 64, 256, 0, stream>>>(cond, refv, refvT, WoT, bo, thw, out);
  attn_ctx_kernel<<<NB * 64, 256, 0, stream>>>(cond, condT, refv, cthw,
                                               out + (size_t)NB * KDIM * DMODEL);
}
